// Head_52252572123299
// MI455X (gfx1250) — matrix operand order repaired and checked by reading, not compiled, not run on hardware
//
#include <hip/hip_runtime.h>

#define DI __device__ __forceinline__

constexpr int Bn = 4, Tn = 2048, Cn = 1024, HSn = 128;

typedef __attribute__((ext_vector_type(16))) __bf16   v16bf;
typedef __attribute__((ext_vector_type(8)))  float    v8f;
typedef __attribute__((ext_vector_type(4)))  unsigned v4u;
typedef __attribute__((ext_vector_type(8)))  int      v8i;
typedef __attribute__((ext_vector_type(4)))  int      v4i;

union FragU { v16bf v; v4u u[2]; };

// float -> bf16 bits, round-to-nearest-even
DI unsigned short f2bf(float f) {
  unsigned u = __builtin_bit_cast(unsigned, f);
  u += 0x7FFFu + ((u >> 16) & 1u);
  return (unsigned short)(u >> 16);
}

// Build a 16-element bf16 fragment from two 16-byte contiguous chunks.
DI v16bf frag_from(const unsigned short* p0, const unsigned short* p1) {
  FragU f;
  f.u[0] = *(const v4u*)p0;
  f.u[1] = *(const v4u*)p1;
  return f.v;
}

DI v8f wmma_bf16(v16bf a, v16bf b, v8f c) {
  return __builtin_amdgcn_wmma_f32_16x16x32_bf16(
      /*neg_a=*/false, a, /*neg_b=*/false, b,
      /*c_mod=*/(short)0, c, /*reuse_a=*/false, /*reuse_b=*/false);
}

// ---------------------------------------------------------------------------
// TDM: async 2D tile load Global -> LDS (CDNA5 Tensor Data Mover).
// D# built per ISA 8.3/8.4: group0 = {count/type, lds_addr, global_addr},
// group1 = {data_size=2B, tensor dims, tile dims, 48-bit dim0 stride}.
// Groups 2/3 zero (2D tensor). Tracked by TENSORcnt.
// 6-arg builtin form (clang-23 / therock-10.0 headers).
// ---------------------------------------------------------------------------
DI void tdm_load_2d(const void* lds_ptr, const void* gptr,
                    unsigned tensor_d0, unsigned tensor_d1,
                    unsigned tile_d0, unsigned tile_d1, unsigned stride_d0) {
  const unsigned lds_off = (unsigned)(size_t)lds_ptr;   // flat->LDS: addr[31:0]
  const unsigned long long ga = (unsigned long long)(size_t)gptr;
  v4u g0;
  g0[0] = 1u;                                            // count=1 (user mode)
  g0[1] = lds_off;                                       // lds_addr [63:32]
  g0[2] = (unsigned)(ga & 0xFFFFFFFFu);                  // global_addr lo
  g0[3] = (unsigned)((ga >> 32) & 0x01FFFFFFu) | (2u << 30); // addr hi | type=2
  v8i g1;
  g1[0] = (int)(1u << 16);                               // wg_mask=0, data_size=2B
  g1[1] = (int)(tensor_d0 << 16);                        // abar=0 | tensor_dim0[15:0]
  g1[2] = (int)((tensor_d0 >> 16) | (tensor_d1 << 16));  // td0 hi | td1 lo
  g1[3] = (int)((tensor_d1 >> 16) | (tile_d0 << 16));    // td1 hi | tile_dim0
  g1[4] = (int)tile_d1;                                  // tile_dim1 | tile_dim2=0
  g1[5] = (int)stride_d0;                                // dim0 stride [31:0]
  g1[6] = 0;                                             // stride hi | dim1 stride lo
  g1[7] = 0;
  v4i z4 = {0, 0, 0, 0};
  v8i z8 = {0, 0, 0, 0, 0, 0, 0, 0};
  __builtin_amdgcn_tensor_load_to_lds(g0, g1, z4, z4, z8, 0);
}

// ---------------------------------------------------------------------------
// Kernel 1: QKV projection.  grid = (8192/64, 3), block = 256 (8 waves).
// which: 0 -> K (row-major), 1 -> Q (row-major, pre-scaled by HS^-0.5),
//        2 -> V (stored transposed: [B][HS][T])
// ---------------------------------------------------------------------------
__global__ __launch_bounds__(256)
void qkv_proj(const float* __restrict__ x,
              const float* __restrict__ Wk,
              const float* __restrict__ Wq,
              const float* __restrict__ Wv,
              unsigned short* __restrict__ Qb,
              unsigned short* __restrict__ Kb,
              unsigned short* __restrict__ Vt)
{
  const int which = blockIdx.y;
  const float* W = (which == 0) ? Wk : (which == 1) ? Wq : Wv;

  const int rowBase = blockIdx.x * 64;
  const int tid  = threadIdx.x;
  const int lane = tid & 31;
  const int wave = tid >> 5;
  const int n    = lane & 15;
  const int half = lane >> 4;
  const int rtile = wave & 3;
  const int cbase = (wave >> 2) * 64;

  __shared__ __align__(16) unsigned short Xs[64][32];
  __shared__ __align__(16) unsigned short Wt[128][32];

  v8f acc[4] = {v8f{}, v8f{}, v8f{}, v8f{}};

  for (int kk = 0; kk < Cn / 32; ++kk) {
    {
      const int li = tid * 8;
      const int r = li >> 5, c = li & 31;
      const float* xs = x + (size_t)(rowBase + r) * Cn + kk * 32 + c;
      float4 a0 = ((const float4*)xs)[0];
      float4 a1 = ((const float4*)xs)[1];
      unsigned* dst = (unsigned*)&Xs[r][c];
      dst[0] = (unsigned)f2bf(a0.x) | ((unsigned)f2bf(a0.y) << 16);
      dst[1] = (unsigned)f2bf(a0.z) | ((unsigned)f2bf(a0.w) << 16);
      dst[2] = (unsigned)f2bf(a1.x) | ((unsigned)f2bf(a1.y) << 16);
      dst[3] = (unsigned)f2bf(a1.z) | ((unsigned)f2bf(a1.w) << 16);
    }
    {
      const int wi = tid * 16;
      const int r = wi >> 7, col = wi & 127;
      const float* ws = W + (size_t)(kk * 32 + r) * HSn + col;
      #pragma unroll
      for (int q4 = 0; q4 < 4; ++q4) {
        float4 a = ((const float4*)ws)[q4];
        Wt[col + q4 * 4 + 0][r] = f2bf(a.x);
        Wt[col + q4 * 4 + 1][r] = f2bf(a.y);
        Wt[col + q4 * 4 + 2][r] = f2bf(a.z);
        Wt[col + q4 * 4 + 3][r] = f2bf(a.w);
      }
    }
    __syncthreads();

    const unsigned short* xr = &Xs[rtile * 16 + n][0];
    v16bf a = frag_from(xr + half * 8, xr + 16 + half * 8);

    #pragma unroll
    for (int ct = 0; ct < 4; ++ct) {
      const unsigned short* wc = &Wt[cbase + ct * 16 + n][0];
      v16bf b = frag_from(wc + 8 * half, wc + 16 + 8 * half);
      acc[ct] = wmma_bf16(a, b, acc[ct]);
    }
    __syncthreads();
  }

  // ---- epilogue: branch on block-uniform `which` ONCE, straight-line stores
  const int r0 = rowBase + rtile * 16 + 8 * half;     // first row of this lane
  if (which == 2) {
    // V: store transposed [B][HS][T]
    const int bi = rowBase / Tn;                       // block-uniform batch
    const int t0 = r0 - bi * Tn;
    #pragma unroll
    for (int ct = 0; ct < 4; ++ct) {
      const int col = cbase + ct * 16 + n;
      unsigned short* vcol = Vt + ((size_t)bi * HSn + col) * Tn + t0;
      #pragma unroll
      for (int i = 0; i < 8; ++i)
        vcol[i] = f2bf(acc[ct][i]);
    }
  } else {
    // K or Q row-major; Q pre-scaled by HS^-0.5
    unsigned short* ob = (which == 1) ? Qb : Kb;
    const float scale = (which == 1) ? 0.08838834764831845f : 1.0f;
    #pragma unroll
    for (int ct = 0; ct < 4; ++ct) {
      const int col = cbase + ct * 16 + n;
      unsigned short* orow = ob + (size_t)r0 * HSn + col;
      #pragma unroll
      for (int i = 0; i < 8; ++i)
        orow[(size_t)i * HSn] = f2bf(acc[ct][i] * scale);
    }
  }
}

// ---------------------------------------------------------------------------
// Kernel 2: flash attention.  grid = (T/64, B), block = 128 (4 waves).
// K and V^T tiles are DMA'd into double-buffered LDS by the TDM (one wave
// issues; all waves consume), overlapping DMA of tile i+1 with WMMA on tile i.
// ---------------------------------------------------------------------------
__global__ __launch_bounds__(128)
void attn(const unsigned short* __restrict__ Qb,
          const unsigned short* __restrict__ Kb,
          const unsigned short* __restrict__ Vt,
          float* __restrict__ out)
{
  const int b    = blockIdx.y;
  const int tid  = threadIdx.x;
  const int lane = tid & 31;
  const int wave = tid >> 5;
  const int n    = lane & 15;
  const int half = lane >> 4;
  const int qrow0 = blockIdx.x * 64 + wave * 16;

  const unsigned short* Qp = Qb + ((size_t)b * Tn + qrow0) * HSn;
  const unsigned short* Kp = Kb + (size_t)b * Tn * HSn;
  const unsigned short* Vp = Vt + (size_t)b * HSn * Tn;

  __shared__ __align__(16) unsigned short Ks[2][32][128];   // K tile (rows x hs)
  __shared__ __align__(16) unsigned short Vs[2][128][32];   // V^T tile (hs x keys)
  __shared__ __align__(16) unsigned short Plds[4][16][32];
  unsigned short (*P)[32] = Plds[wave];

  // Q fragments (A-layout): lane holds row M = n, K split per ISA table
  v16bf qf[4];
  {
    const unsigned short* qr = Qp + (size_t)n * HSn;
    #pragma unroll
    for (int c = 0; c < 4; ++c)
      qf[c] = frag_from(qr + c * 32 + half * 8, qr + c * 32 + 16 + half * 8);
  }

  v8f o[8];
  #pragma unroll
  for (int ct = 0; ct < 8; ++ct) o[ct] = v8f{};
  float m_[8], l_[8];
  #pragma unroll
  for (int i = 0; i < 8; ++i) { m_[i] = -3.0e38f; l_[i] = 0.0f; }

  constexpr float LOG2E = 1.4426950408889634f;

  // Prologue: DMA tile 0 (K: 32x128 row-stride 128; V^T: 128x32 row-stride T)
  if (wave == 0) {
    tdm_load_2d(&Ks[0][0][0], Kp,       128, 32, 128, 32, 128);
    tdm_load_2d(&Vs[0][0][0], Vp,        32, 128, 32, 128, Tn);
  }

  int buf = 0;
  for (int kb = 0; kb < Tn; kb += 32) {
    if (wave == 0) {
      if (kb + 32 < Tn) {
        tdm_load_2d(&Ks[buf ^ 1][0][0], Kp + (size_t)(kb + 32) * HSn,
                    128, 32, 128, 32, 128);
        tdm_load_2d(&Vs[buf ^ 1][0][0], Vp + (kb + 32),
                    32, 128, 32, 128, Tn);
        __builtin_amdgcn_s_wait_tensorcnt(2);   // tile 'buf' complete
      } else {
        __builtin_amdgcn_s_wait_tensorcnt(0);
      }
    }
    __syncthreads();   // LDS tile visible to all waves

    // ---- S = Q * K^T (16x32), two 16x16 tiles, from LDS ------------------
    v8f s0 = v8f{}, s1 = v8f{};
    #pragma unroll
    for (int c = 0; c < 4; ++c) {
      const unsigned short* kr0 = &Ks[buf][n][c * 32 + 8 * half];
      s0 = wmma_bf16(qf[c], frag_from(kr0, kr0 + 16), s0);
      const unsigned short* kr1 = &Ks[buf][16 + n][c * 32 + 8 * half];
      s1 = wmma_bf16(qf[c], frag_from(kr1, kr1 + 16), s1);
    }

    // ---- online softmax over 32 keys; stage P(bf16) to LDS ---------------
    #pragma unroll
    for (int i = 0; i < 8; ++i) {
      float t = fmaxf(s0[i], s1[i]);
      t = fmaxf(t, __shfl_xor(t, 1, 32));
      t = fmaxf(t, __shfl_xor(t, 2, 32));
      t = fmaxf(t, __shfl_xor(t, 4, 32));
      t = fmaxf(t, __shfl_xor(t, 8, 32));
      const float mn    = fmaxf(m_[i], t);
      const float alpha = exp2f((m_[i] - mn) * LOG2E);
      const float p0    = exp2f((s0[i] - mn) * LOG2E);
      const float p1    = exp2f((s1[i] - mn) * LOG2E);
      float rs = p0 + p1;
      rs += __shfl_xor(rs, 1, 32);
      rs += __shfl_xor(rs, 2, 32);
      rs += __shfl_xor(rs, 4, 32);
      rs += __shfl_xor(rs, 8, 32);
      l_[i] = l_[i] * alpha + rs;
      m_[i] = mn;
      P[i + 8 * half][n]      = f2bf(p0);
      P[i + 8 * half][16 + n] = f2bf(p1);
      #pragma unroll
      for (int ct = 0; ct < 8; ++ct) o[ct][i] *= alpha;
    }
    asm volatile("s_wait_dscnt 0" ::: "memory");   // within-wave LDS RAW

    // ---- P fragment (A-layout) -------------------------------------------
    const unsigned short* pr = &P[n][0];
    v16bf pf = frag_from(pr + half * 8, pr + 16 + half * 8);

    // ---- O += P * V : 8 col tiles from LDS V^T ---------------------------
    #pragma unroll
    for (int ct = 0; ct < 8; ++ct) {
      const unsigned short* vr = &Vs[buf][ct * 16 + n][8 * half];
      o[ct] = wmma_bf16(pf, frag_from(vr, vr + 16), o[ct]);
    }

    __syncthreads();   // all waves done with 'buf' before TDM overwrites it
    buf ^= 1;
  }

  // ---- epilogue: normalize by l and store f32 -----------------------------
  float* orow = out + ((size_t)b * Tn + qrow0) * HSn;
  #pragma unroll
  for (int i = 0; i < 8; ++i) {
    const float inv = 1.0f / l_[i];
    #pragma unroll
    for (int ct = 0; ct < 8; ++ct)
      orow[(size_t)(i + 8 * half) * HSn + ct * 16 + n] = o[ct][i] * inv;
  }
}

// ---------------------------------------------------------------------------
extern "C" void kernel_launch(void* const* d_in, const int* in_sizes, int n_in,
                              void* d_out, int out_size, void* d_ws, size_t ws_size,
                              hipStream_t stream) {
  const float* x  = (const float*)d_in[0];
  const float* Wk = (const float*)d_in[1];
  const float* Wq = (const float*)d_in[2];
  const float* Wv = (const float*)d_in[3];

  const size_t per = (size_t)Bn * Tn * HSn;
  unsigned short* Qb = (unsigned short*)d_ws;
  unsigned short* Kb = Qb + per;
  unsigned short* Vt = Kb + per;                      // stored [B][HS][T]
  float* out = (float*)d_out;

  dim3 g1((Bn * Tn) / 64, 3);
  qkv_proj<<<g1, 256, 0, stream>>>(x, Wk, Wq, Wv, Qb, Kb, Vt);

  dim3 g2(Tn / 64, Bn);
  attn<<<g2, 128, 0, stream>>>(Qb, Kb, Vt, out);
}
